// TinyTransformer_17575006175754
// MI455X (gfx1250) — compile-verified, matrix operand order checked by reading, not run on hardware
//
#include <hip/hip_runtime.h>
#include <cmath>

typedef unsigned short u16;
typedef __bf16 bf16t;
typedef bf16t v16bf __attribute__((ext_vector_type(16)));
typedef float  v8f   __attribute__((ext_vector_type(8)));

#define GEMM_RELU 2

__device__ __forceinline__ u16 f2b(float f) {
  unsigned u = __float_as_uint(f);
  u += 0x7fffu + ((u >> 16) & 1u);           // round-to-nearest-even
  return (u16)(u >> 16);
}

// Copy 16B global -> LDS via the CDNA5 async LDS-DMA path (tracked by ASYNCcnt).
// vdst operand = wave-relative LDS byte offset = low 32 bits of the generic
// pointer (generic LDS addresses are {aperture_hi, lds_offset}).
__device__ __forceinline__ void copy16_g2l(const u16* gsrc, u16* ldst) {
  unsigned lds_off = (unsigned)(unsigned long long)ldst;
  unsigned long long ga = (unsigned long long)gsrc;
  asm volatile("global_load_async_to_lds_b128 %0, %1, off"
               :: "v"(lds_off), "v"(ga)
               : "memory");
}

__device__ __forceinline__ void wait_async0() {
#if __has_builtin(__builtin_amdgcn_s_wait_asynccnt)
  __builtin_amdgcn_s_wait_asynccnt(0);
#else
  asm volatile("s_wait_asynccnt 0x0" ::: "memory");
#endif
}

union FragU { uint4 u4[2]; v16bf v; };

// Per-lane 16xbf16 fragment gather from an LDS row (A/B 16-bit layout:
// elems 0..7 = K(0..7)+khalf*8, elems 8..15 = K(16..23)+khalf*8).
__device__ __forceinline__ v16bf load_frag(const u16* row, int khalf) {
  const uint4* p = (const uint4*)(row + khalf * 8);
  FragU f;
  f.u4[0] = p[0];
  f.u4[1] = p[2];
  return f.v;
}

// ---------------------------------------------------------------------------
// Batched bf16 GEMM, B pre-transposed ([N][K], ldb = K-stride).
// Block tile 128x128x32, 8 waves as 4(M)x2(N), wave tile 32x64 (8 WMMA/step).
// Double-buffered LDS, async global->LDS staging pipelined against WMMA.
// C = act(alpha * A*Bt^T + bias). Batch z -> (z/nb2, z%nb2).
// ---------------------------------------------------------------------------
__global__ __launch_bounds__(256) void gemm_bt_wmma(
    const u16* __restrict__ A, const u16* __restrict__ Bt,
    const float* __restrict__ bias, float* __restrict__ Cf, u16* __restrict__ Cb,
    int K, int lda, int ldb, int ldc,
    long long aB1, long long aB2, long long bB1, long long bB2,
    long long cB1, long long cB2, long long biasB2,
    int nb2, float alpha, int flags)
{
  constexpr int BM = 128, BN = 128, BK = 32, LS = 40; // LS = padded LDS row stride
  __shared__ __align__(16) u16 As[2 * BM * LS];
  __shared__ __align__(16) u16 Bs[2 * BN * LS];

  const int tid    = threadIdx.x;
  const int wave   = tid >> 5;
  const int lane   = tid & 31;
  const int lm     = lane & 15;
  const int khalf  = lane >> 4;
  const int wave_m = wave >> 1;   // 0..3
  const int wave_n = wave & 1;    // 0..1

  const int z  = blockIdx.z;
  const int z1 = z / nb2;
  const int z2 = z - z1 * nb2;
  A  += (long long)z1 * aB1 + (long long)z2 * aB2;
  Bt += (long long)z1 * bB1 + (long long)z2 * bB2;
  const long long cOff = (long long)z1 * cB1 + (long long)z2 * cB2;
  if (bias) bias += (long long)z2 * biasB2;

  const int m0 = blockIdx.y * BM;
  const int n0 = blockIdx.x * BN;

  v8f acc[2][4];
  for (int a = 0; a < 2; ++a)
    for (int j = 0; j < 4; ++j)
      for (int r = 0; r < 8; ++r) acc[a][j][r] = 0.0f;

  auto stage = [&](int kt, int bufi) {
    u16* Ad = &As[bufi * BM * LS];
    u16* Bd = &Bs[bufi * BN * LS];
    for (int r = 0; r < 2; ++r) {            // A: 128x32 = 512 16B chunks
      int i = tid + r * 256;
      int m = i >> 2, k8 = i & 3;
      copy16_g2l(A + (long long)(m0 + m) * lda + kt + k8 * 8,
                 Ad + m * LS + k8 * 8);
    }
    for (int r = 0; r < 2; ++r) {            // Bt: 128x32 = 512 16B chunks
      int i = tid + r * 256;
      int n = i >> 2, k8 = i & 3;
      copy16_g2l(Bt + (long long)(n0 + n) * ldb + kt + k8 * 8,
                 Bd + n * LS + k8 * 8);
    }
  };

  const int nT = K / BK;
  stage(0, 0);
  for (int t = 0; t < nT; ++t) {
    wait_async0();
    __syncthreads();                          // buf t ready for all waves
    if (t + 1 < nT) stage((t + 1) * BK, (t + 1) & 1); // DMA next tile during WMMA
    const u16* Ab = &As[(t & 1) * BM * LS];
    const u16* Bb = &Bs[(t & 1) * BN * LS];
    v16bf a0 = load_frag(&Ab[(wave_m * 32 + lm) * LS], khalf);
    v16bf a1 = load_frag(&Ab[(wave_m * 32 + 16 + lm) * LS], khalf);
    for (int j = 0; j < 4; ++j) {
      v16bf bfrag = load_frag(&Bb[(wave_n * 64 + j * 16 + lm) * LS], khalf);
      acc[0][j] = __builtin_amdgcn_wmma_f32_16x16x32_bf16(
          false, a0, false, bfrag, (short)0, acc[0][j], false, false);
      acc[1][j] = __builtin_amdgcn_wmma_f32_16x16x32_bf16(
          false, a1, false, bfrag, (short)0, acc[1][j], false, false);
    }
    __syncthreads();                          // reads done before buf reuse
  }

  for (int a = 0; a < 2; ++a)
    for (int j = 0; j < 4; ++j) {
      int gn = n0 + wave_n * 64 + j * 16 + lm;
      float bv = bias ? bias[gn] : 0.0f;
      for (int r = 0; r < 8; ++r) {
        int gm = m0 + wave_m * 32 + a * 16 + r + khalf * 8;
        float v = acc[a][j][r] * alpha + bv;
        if (flags & GEMM_RELU) v = fmaxf(v, 0.0f);
        long long idx = cOff + (long long)gm * ldc + gn;
        if (Cf) Cf[idx] = v;
        if (Cb) Cb[idx] = f2b(v);
      }
    }
}

// ---------------------------------------------------------------------------
// KtV = k^T @ v per (b,h): 128x128, K=1024. Both operands need a transform,
// so stage with scalar gathers (coalesced along the fast dim). Stores result
// TRANSPOSED (KTVT[f][e]) so the q@(KtV) GEMM can use the BT fast path.
// ---------------------------------------------------------------------------
__global__ __launch_bounds__(256) void gemm_ktv(
    const u16* __restrict__ qkv, u16* __restrict__ KTVT)
{
  constexpr int LS = 40;
  __shared__ __align__(16) u16 As[128 * LS];
  __shared__ __align__(16) u16 Bs[64 * LS];

  const int tid   = threadIdx.x;
  const int wave  = tid >> 5;
  const int lm    = tid & 15;
  const int khalf = (tid & 31) >> 4;
  const int z  = blockIdx.z;
  const int n0 = blockIdx.x * 64;

  const u16* Kp = qkv + (long long)z * 393216 + 128;
  const u16* Vp = qkv + (long long)z * 393216 + 256;

  v8f acc[4];
  for (int j = 0; j < 4; ++j)
    for (int r = 0; r < 8; ++r) acc[j][r] = 0.0f;

  for (int kt = 0; kt < 1024; kt += 32) {
    for (int i = tid; i < 128 * 32; i += 256) {     // As[m=e][kk=t] = k[t][e]
      int m = i & 127, kk = i >> 7;
      As[m * LS + kk] = Kp[(long long)(kt + kk) * 384 + m];
    }
    for (int i = tid; i < 64 * 32; i += 256) {      // Bs[n=f][kk=t] = v[t][f]
      int n = i & 63, kk = i >> 6;
      Bs[n * LS + kk] = Vp[(long long)(kt + kk) * 384 + n0 + n];
    }
    __syncthreads();
    v16bf af = load_frag(&As[(wave * 16 + lm) * LS], khalf);
    for (int j = 0; j < 4; ++j) {
      v16bf bfrag = load_frag(&Bs[(j * 16 + lm) * LS], khalf);
      acc[j] = __builtin_amdgcn_wmma_f32_16x16x32_bf16(
          false, af, false, bfrag, (short)0, acc[j], false, false);
    }
    __syncthreads();
  }

  for (int j = 0; j < 4; ++j)
    for (int r = 0; r < 8; ++r) {
      int gm = wave * 16 + r + khalf * 8;           // e
      int gn = n0 + j * 16 + lm;                    // f
      KTVT[(long long)z * 16384 + (long long)gn * 128 + gm] = f2b(acc[j][r]);
    }
}

// ---------------------------------------------------------------------------
__global__ void cvtT_f32_bf16(const float* __restrict__ in, u16* __restrict__ out,
                              int Kd, int Nd) {
  long long base = (long long)blockIdx.y * Kd * Nd;
  int idx = blockIdx.x * 256 + threadIdx.x;
  if (idx < Kd * Nd) {
    int n = idx / Kd, k = idx - n * Kd;
    out[base + idx] = f2b(in[base + (long long)k * Nd + n]); // out[n][k] = in[k][n]
  }
}

__global__ __launch_bounds__(128) void embed_kernel(
    const int* __restrict__ inputs, const float* __restrict__ emb,
    float* __restrict__ Xf, u16* __restrict__ Xb)
{
  const int bs  = blockIdx.x;
  const int s   = bs & 1023;
  const int tok = inputs[bs];
  for (int i = 0; i < 4; ++i) {
    int col = threadIdx.x + i * 128;
    float e = emb[(long long)tok * 512 + col];
    float expo = (2.0f * (float)(col >> 1)) * (1.0f / 512.0f);
    float ang = (float)s / __powf(10000.0f, expo);
    float p = (col & 1) ? __cosf(ang) : __sinf(ang);
    float v = e + p;
    long long idx = (long long)bs * 512 + col;
    Xf[idx] = v;
    Xb[idx] = f2b(v);
  }
}

__global__ __launch_bounds__(128) void ln_kernel(
    const float* __restrict__ in, const float* __restrict__ res,
    const float* __restrict__ g, const float* __restrict__ beta,
    float* __restrict__ outF, u16* __restrict__ outB)
{
  __shared__ float red[128];
  const long long row = blockIdx.x;
  const int tid = threadIdx.x;
  float loc[4]; float s = 0.f;
  for (int i = 0; i < 4; ++i) {
    int col = tid + i * 128;
    float v = in[row * 512 + col] + res[row * 512 + col];
    loc[i] = v; s += v;
  }
  red[tid] = s; __syncthreads();
  for (int o = 64; o > 0; o >>= 1) { if (tid < o) red[tid] += red[tid + o]; __syncthreads(); }
  float mu = red[0] * (1.0f / 512.0f);
  __syncthreads();
  float s2 = 0.f;
  for (int i = 0; i < 4; ++i) { float d = loc[i] - mu; s2 += d * d; }
  red[tid] = s2; __syncthreads();
  for (int o = 64; o > 0; o >>= 1) { if (tid < o) red[tid] += red[tid + o]; __syncthreads(); }
  float rstd = rsqrtf(red[0] * (1.0f / 512.0f) + 0.001f);
  for (int i = 0; i < 4; ++i) {
    int col = tid + i * 128;
    float v = (loc[i] - mu) * rstd * g[col] + beta[col];
    outF[row * 512 + col] = v;
    if (outB) outB[row * 512 + col] = f2b(v);
  }
}

__global__ __launch_bounds__(512) void head_kernel(
    const float* __restrict__ X, const float* __restrict__ Wh,
    const float* __restrict__ bh, const float* __restrict__ Wf,
    const float* __restrict__ bfin, float* __restrict__ out)
{
  __shared__ float x0[512];
  __shared__ float red[512];
  const int b = blockIdx.x;
  const int tid = threadIdx.x;
  x0[tid] = X[(long long)b * 1024 * 512 + tid];   // row s = 0 of batch b
  __syncthreads();
  float acc = bh[tid];
  for (int d = 0; d < 512; ++d) acc = fmaf(x0[d], Wh[d * 512 + tid], acc);
  float h = fmaxf(acc, 0.0f);
  red[tid] = h * Wf[tid];
  __syncthreads();
  for (int o = 256; o > 0; o >>= 1) { if (tid < o) red[tid] += red[tid + o]; __syncthreads(); }
  if (tid == 0) {
    float logit = red[0] + bfin[0];
    out[b] = logit;
    out[32 + b] = 1.0f / (1.0f + expf(-logit));
  }
}

// ---------------------------------------------------------------------------
extern "C" void kernel_launch(void* const* d_in, const int* in_sizes, int n_in,
                              void* d_out, int out_size, void* d_ws, size_t ws_size,
                              hipStream_t stream)
{
  const int*   inputs = (const int*)  d_in[0];
  const float* emb    = (const float*)d_in[1];
  const float* Wp     = (const float*)d_in[2];
  const float* bp     = (const float*)d_in[3];
  const float* Wo     = (const float*)d_in[4];
  const float* bo     = (const float*)d_in[5];
  const float* ln1g   = (const float*)d_in[6];
  const float* ln1b   = (const float*)d_in[7];
  const float* W1     = (const float*)d_in[8];
  const float* b1     = (const float*)d_in[9];
  const float* W2     = (const float*)d_in[10];
  const float* b2     = (const float*)d_in[11];
  const float* ln2g   = (const float*)d_in[12];
  const float* ln2b   = (const float*)d_in[13];
  const float* Wh     = (const float*)d_in[14];
  const float* bh     = (const float*)d_in[15];
  const float* Wf     = (const float*)d_in[16];
  const float* bfin   = (const float*)d_in[17];
  float* out = (float*)d_out;

  // Workspace sub-allocation (all sizes multiples of 256B).
  char* p = (char*)d_ws;
  u16*   WpT  = (u16*)p;   p += (size_t)786432   * 2;  // [h][384][512]
  u16*   WoT  = (u16*)p;   p += (size_t)262144   * 2;  // [512][512]
  u16*   W1T  = (u16*)p;   p += (size_t)1048576  * 2;  // [2048][512]
  u16*   W2T  = (u16*)p;   p += (size_t)1048576  * 2;  // [512][2048]
  float* X0f  = (float*)p; p += (size_t)16777216 * 4;  // B*S*D residual
  u16*   X0b  = (u16*)p;   p += (size_t)16777216 * 2;
  u16*   QKV  = (u16*)p;   p += (size_t)50331648 * 2;  // [b,h,s,384]
  u16*   KTVT = (u16*)p;   p += (size_t)2097152  * 2;  // [b,h][f][e] (transposed)
  u16*   CONC = (u16*)p;   p += (size_t)16777216 * 2;  // [b,s,512]
  float* ACT  = (float*)p; p += (size_t)16777216 * 4;  // attn out / ffn2 out
  float* X1f  = (float*)p; p += (size_t)16777216 * 4;
  u16*   X1b  = (u16*)p;   p += (size_t)16777216 * 2;
  u16*   F1   = (u16*)p;   p += (size_t)67108864 * 2;  // B*S*FFN
  float* X2f  = X0f;                                   // reuse residual buffer

  // Weights -> transposed bf16 ([N][K])
  cvtT_f32_bf16<<<dim3(768,  4), 256, 0, stream>>>(Wp, WpT, 512, 384);
  cvtT_f32_bf16<<<dim3(1024, 1), 256, 0, stream>>>(Wo, WoT, 512, 512);
  cvtT_f32_bf16<<<dim3(4096, 1), 256, 0, stream>>>(W1, W1T, 512, 2048);
  cvtT_f32_bf16<<<dim3(4096, 1), 256, 0, stream>>>(W2, W2T, 2048, 512);

  // x = emb[inputs] + pos
  embed_kernel<<<32 * 1024, 128, 0, stream>>>(inputs, emb, X0f, X0b);

  // QKV projection: per (b,h): (1024x512)@(512x384) + bp[h]
  gemm_bt_wmma<<<dim3(3, 8, 128), 256, 0, stream>>>(
      X0b, WpT, bp, nullptr, QKV,
      512, 512, 512, 384,
      524288LL, 0LL, 0LL, 196608LL,
      1572864LL, 393216LL, 384LL,
      4, 1.0f, 0);

  // Linear attention: KTVT = (k^T @ v)^T per (b,h)
  gemm_ktv<<<dim3(2, 1, 128), 256, 0, stream>>>(QKV, KTVT);

  // heads = scale * q @ KtV -> concat layout [b,s,h*128+e]
  gemm_bt_wmma<<<dim3(1, 8, 128), 256, 0, stream>>>(
      QKV, KTVT, nullptr, nullptr, CONC,
      128, 384, 128, 512,
      1572864LL, 393216LL, 65536LL, 16384LL,
      524288LL, 128LL, 0LL,
      4, 0.03125f, 0);

  // attn = concat @ Wo + bo   (32768x512, K=512)
  gemm_bt_wmma<<<dim3(4, 256, 1), 256, 0, stream>>>(
      CONC, WoT, bo, ACT, nullptr,
      512, 512, 512, 512,
      0LL, 0LL, 0LL, 0LL, 0LL, 0LL, 0LL,
      1, 1.0f, 0);

  // x1 = LN(attn + x0)
  ln_kernel<<<32768, 128, 0, stream>>>(ACT, X0f, ln1g, ln1b, X1f, X1b);

  // FFN1: relu(x1 @ W1 + b1)   (32768x2048, K=512)
  gemm_bt_wmma<<<dim3(16, 256, 1), 256, 0, stream>>>(
      X1b, W1T, b1, nullptr, F1,
      512, 512, 512, 2048,
      0LL, 0LL, 0LL, 0LL, 0LL, 0LL, 0LL,
      1, 1.0f, GEMM_RELU);

  // FFN2: h = ffn1 @ W2 + b2   (32768x512, K=2048)
  gemm_bt_wmma<<<dim3(4, 256, 1), 256, 0, stream>>>(
      F1, W2T, b2, ACT, nullptr,
      2048, 2048, 2048, 512,
      0LL, 0LL, 0LL, 0LL, 0LL, 0LL, 0LL,
      1, 1.0f, 0);

  // x2 = LN(h + x1)
  ln_kernel<<<32768, 128, 0, stream>>>(ACT, X1f, ln2g, ln2b, X2f, nullptr);

  // classifier head on x2[:, 0]
  head_kernel<<<32, 512, 0, stream>>>(X2f, Wh, bh, Wf, bfin, out);
}
